// ScaledDotProductAttention_14551349199520
// MI455X (gfx1250) — matrix operand order repaired and checked by reading, not compiled, not run on hardware
//
#include <hip/hip_runtime.h>
#include <cmath>
#include <type_traits>

typedef __attribute__((ext_vector_type(16))) _Float16 v16h;
typedef __attribute__((ext_vector_type(8)))  _Float16 v8h;
typedef __attribute__((ext_vector_type(4)))  _Float16 v4h;
typedef __attribute__((ext_vector_type(8)))  float    v8f;
typedef __attribute__((ext_vector_type(4)))  float    v4f;

#define BATCH  4
#define HEADS  16
#define SEQ    2048
#define DIM    64
#define QTILE  64     // query rows per workgroup
#define KTILE  32     // keys per inner iteration
#define NWAVES 4

// LDS pitches chosen so every fragment read is 16B/32B aligned
#define KPITCH 72     // Ks row: 72 halfs = 144B (32B-aligned sub-chunks at 0/16/32/48 halfs)
#define VPITCH 48     // Vs row: 48 halfs = 96B  (32B-aligned sub-chunks at 0/16 halfs)
#define PPITCH 40     // Ps row: 40 halfs = 80B  (16B-aligned sub-chunks at 0/8/16/24 halfs)

// Raw v_max_num_f32: ISA maximumNumber already quiets NaNs; avoids LLVM's
// sNaN canonicalization.
__device__ __forceinline__ float fmax_raw(float a, float b) {
  float r;
  asm("v_max_num_f32 %0, %1, %2" : "=v"(r) : "v"(a), "v"(b));
  return r;
}

// Raw v_exp_f32: flushes deeply-negative inputs to 0 (wanted for masked p),
// avoids the 6-op denormal-guard expansion of exp2f.
#if __has_builtin(__builtin_amdgcn_exp2f)
__device__ __forceinline__ float exp2_raw(float x) { return __builtin_amdgcn_exp2f(x); }
#else
__device__ __forceinline__ float exp2_raw(float x) { return exp2f(x); }
#endif

// 16-lane xor-butterfly max on the VALU. Inline-asm v_permlane16_b32: dest is
// untied (EXEC all-ones -> 'old' never consumed), killing the per-step v_mov
// the builtin forces.
__device__ __forceinline__ float plane16(float x, unsigned s0, unsigned s1) {
  float r;
  asm("v_permlane16_b32 %0, %1, %2, %3" : "=v"(r) : "v"(x), "s"(s0), "s"(s1));
  return r;
}
__device__ __forceinline__ float rowmax16(float x) {
  x = fmax_raw(x, plane16(x, 0x67452301u, 0xEFCDAB89u));  // xor 1
  x = fmax_raw(x, plane16(x, 0x54761032u, 0xDCFE98BAu));  // xor 2
  x = fmax_raw(x, plane16(x, 0x32107654u, 0xBA98FEDCu));  // xor 4
  x = fmax_raw(x, plane16(x, 0xFEDCBA98u, 0x76543210u));  // xor 8
  return x;
}

__global__ __launch_bounds__(128) void fattn_causal_f16wmma(
    const float* __restrict__ Qg, const float* __restrict__ Kg,
    const float* __restrict__ Vg, float* __restrict__ Og)
{
  const int tid  = threadIdx.x;
  const int wid  = tid >> 5;          // wave id 0..3
  const int lane = tid & 31;
  const int lo   = lane & 15;
  const int hi8  = (lane >> 4) * 8;   // +8 rows for upper half-wave
  const int hi16 = (lane >> 4) * 16;  // +16 K for upper half-wave

  const int q0 = blockIdx.x * QTILE;
  const int bh = blockIdx.y;
  const size_t base = (size_t)bh * SEQ * DIM;

  // double-buffered K/V tiles; per-wave P relayout tile
  __shared__ _Float16 Ks[2][KTILE][KPITCH];     // Ks[buf][key][d]  (row-major)
  __shared__ _Float16 Vs[2][DIM][VPITCH];       // Vs[buf][d][key]  (transposed)
  __shared__ _Float16 Ps[NWAVES][16][PPITCH];   // per-wave P tile, [row][key]

  // 1/sqrt(64) * log2(e), folded into Q so QK^T emerges pre-scaled in log2 domain
  const float SSCL = 0.125f * 1.4426950408889634f;

  // ---- Q A-fragments (16x32 f16, two d-chunks), loaded once, pre-scaled ----
  // A layout: lane(0..15)->row M=lane; half i -> K = hi16 + i
  const int qrow = q0 + wid * 16 + lo;
  v16h aq[2];
  #pragma unroll
  for (int c = 0; c < 2; ++c) {
    #pragma unroll
    for (int i = 0; i < 16; ++i) {
      const int dk = c * 32 + hi16 + i;
      aq[c][i] = (_Float16)(Qg[base + (size_t)qrow * DIM + dk] * SSCL);
    }
  }

  // all-ones B fragment: rowsum(P) = P(16x32) x ones(32x16) via one extra WMMA
  v16h ones_b;
  #pragma unroll
  for (int i = 0; i < 16; ++i) ones_b[i] = (_Float16)1.0f;

  v8f acc[4] = {};                      // O accumulators: 4 d-chunks of 16x16 f32
  float mrow[8], lrow[8], sfv[8];
  #pragma unroll
  for (int r = 0; r < 8; ++r) { mrow[r] = -1e30f; lrow[r] = 0.0f; }

  // ---- register-staged global->LDS pipeline (one barrier per block) ----
  v4f kreg[4], vreg[4];
  auto load_regs = [&](int kb) {        // issue global_load_b128s, no wait yet
    #pragma unroll
    for (int j = 0; j < 4; ++j) {
      const int e4 = tid + j * 128;     // 512 float4's = 32x64 elements
      const int ky = e4 >> 4;           // 0..31
      const int d  = (e4 & 15) * 4;     // 0..60
      const size_t g = base + (size_t)(kb + ky) * DIM + d;
      kreg[j] = *(const v4f*)&Kg[g];
      vreg[j] = *(const v4f*)&Vg[g];
    }
  };
  auto store_lds = [&](int buf) {       // cvt f32->f16 and commit to LDS
    #pragma unroll
    for (int j = 0; j < 4; ++j) {
      const int e4 = tid + j * 128;
      const int ky = e4 >> 4;
      const int d  = (e4 & 15) * 4;
      *(v4h*)&Ks[buf][ky][d] = __builtin_convertvector(kreg[j], v4h);
      const v4h vh = __builtin_convertvector(vreg[j], v4h);
      Vs[buf][d + 0][ky] = vh[0];
      Vs[buf][d + 1][ky] = vh[1];
      Vs[buf][d + 2][ky] = vh[2];
      Vs[buf][d + 3][ky] = vh[3];
    }
  };

  // One 32-key block of math. DOMASK compile-time; diagonal blocks also take
  // a wave-uniform early-out (EXEC stays all-ones for WMMA inside).
  auto compute = [&](int buf, int kb, auto domask_c) {
    constexpr bool DOMASK = decltype(domask_c)::value;
    if (DOMASK && kb > q0 + wid * 16 + 15) return;   // wave-uniform skip

    // ---- scores = Q * K^T : two 16(key)-wide tiles, accumulate over d ----
    v8f s0 = {}, s1 = {};
    {
      const v16h bk00 = *(const v16h*)&Ks[buf][lo     ][     hi16];
      const v16h bk01 = *(const v16h*)&Ks[buf][lo     ][32 + hi16];
      const v16h bk10 = *(const v16h*)&Ks[buf][16 + lo][     hi16];
      const v16h bk11 = *(const v16h*)&Ks[buf][16 + lo][32 + hi16];
      s0 = __builtin_amdgcn_wmma_f32_16x16x32_f16(false, aq[0], false, bk00, (short)0, s0, false, false);
      s0 = __builtin_amdgcn_wmma_f32_16x16x32_f16(false, aq[1], false, bk01, (short)0, s0, false, false);
      s1 = __builtin_amdgcn_wmma_f32_16x16x32_f16(false, aq[0], false, bk10, (short)0, s1, false, false);
      s1 = __builtin_amdgcn_wmma_f32_16x16x32_f16(false, aq[1], false, bk11, (short)0, s1, false, false);
    }

    // ---- online softmax (max + exp only; sums via WMMA below) ----
    const int k0g = kb + lo;
    const int k1g = k0g + 16;
    #pragma unroll
    for (int r = 0; r < 8; ++r) {
      float ms0 = s0[r], ms1 = s1[r];
      if (DOMASK) {
        const int qr = q0 + wid * 16 + r + hi8;
        ms0 = (k0g <= qr) ? ms0 : -1e30f;
        ms1 = (k1g <= qr) ? ms1 : -1e30f;
      }
      const float rm = rowmax16(fmax_raw(ms0, ms1));
      const float mn = fmax_raw(mrow[r], rm);
      sfv[r] = exp2_raw(mrow[r] - mn);
      mrow[r] = mn;

      Ps[wid][r + hi8][lo]      = (_Float16)exp2_raw(ms0 - mn);
      Ps[wid][r + hi8][16 + lo] = (_Float16)exp2_raw(ms1 - mn);
    }

    // ---- relayout P (C-layout -> A-layout) through per-wave LDS ----
    const v8h plo = *(const v8h*)&Ps[wid][lo][hi16];
    const v8h phi = *(const v8h*)&Ps[wid][lo][hi16 + 8];
    const v16h ap = __builtin_shufflevector(plo, phi,
        0,1,2,3,4,5,6,7,8,9,10,11,12,13,14,15);

    // ---- row sums: one WMMA vs all-ones B; D[r] = rowsum(row r+hi8) ----
    v8f rsv = {};
    rsv = __builtin_amdgcn_wmma_f32_16x16x32_f16(false, ap, false, ones_b, (short)0, rsv, false, false);

    #pragma unroll
    for (int r = 0; r < 8; ++r) {
      lrow[r] = lrow[r] * sfv[r] + rsv[r];
      #pragma unroll
      for (int c = 0; c < 4; ++c) acc[c][r] *= sfv[r];
    }

    // ---- O += P * V ----
    #pragma unroll
    for (int c = 0; c < 4; ++c) {
      const v16h bv = *(const v16h*)&Vs[buf][c * 16 + lo][hi16];
      acc[c] = __builtin_amdgcn_wmma_f32_16x16x32_f16(false, ap, false, bv, (short)0, acc[c], false, false);
    }
  };

  // ---- pipelined main loop, split interior/diagonal so each hot loop has
  //      exactly one compute instance (no per-iteration register shuffles) ----
  const int nblocks = (q0 + QTILE) / KTILE;
  const int nfull   = q0 / KTILE;       // blocks needing no causal mask

  load_regs(0);
  store_lds(0);
  int cur = 0;

  // interior blocks: no masking at all
  for (int ib = 0; ib < nfull; ++ib) {
    const int kb = ib * KTILE;
    __syncthreads();                    // buf 'cur' staged; buf 'cur^1' free
    load_regs(kb + KTILE);              // always a next block here
    if (ib + 2 < nblocks) {             // warm L2 two blocks ahead
      const size_t ng = base + (size_t)(kb + 2 * KTILE) * DIM + tid * 16;
      __builtin_prefetch(&Kg[ng], 0, 3);
      __builtin_prefetch(&Vg[ng], 0, 3);
    }
    compute(cur, kb, std::integral_constant<bool, false>{});
    store_lds(cur ^ 1);                 // waits loads only here
    cur ^= 1;
  }
  // diagonal blocks (always 2): causal masking + per-wave early-out
  for (int ib = nfull; ib < nblocks; ++ib) {
    const int kb = ib * KTILE;
    __syncthreads();
    const bool haveNext = (ib + 1 < nblocks);
    if (haveNext) load_regs(kb + KTILE);
    compute(cur, kb, std::integral_constant<bool, true>{});
    if (haveNext) store_lds(cur ^ 1);
    cur ^= 1;
  }

  // ---- epilogue: normalize and store ----
  #pragma unroll
  for (int r = 0; r < 8; ++r) {
    const int qr = q0 + wid * 16 + r + hi8;
    const float inv = 1.0f / lrow[r];
    #pragma unroll
    for (int c = 0; c < 4; ++c)
      Og[base + (size_t)qr * DIM + c * 16 + lo] = acc[c][r] * inv;
  }
}

extern "C" void kernel_launch(void* const* d_in, const int* in_sizes, int n_in,
                              void* d_out, int out_size, void* d_ws, size_t ws_size,
                              hipStream_t stream) {
  (void)in_sizes; (void)n_in; (void)out_size; (void)d_ws; (void)ws_size;
  const float* Q = (const float*)d_in[0];
  const float* K = (const float*)d_in[1];
  const float* V = (const float*)d_in[2];
  // d_in[3] is the causal mask; causality is applied analytically in-kernel.
  float* O = (float*)d_out;

  dim3 grid(SEQ / QTILE, BATCH * HEADS);
  dim3 block(128);
  fattn_causal_f16wmma<<<grid, block, 0, stream>>>(Q, K, V, O);
}
